// ShardedGptSelfAttention_62680752717959
// MI455X (gfx1250) — matrix operand order repaired and checked by reading, not compiled, not run on hardware
//
#include <hip/hip_runtime.h>
#include <hip/hip_bf16.h>

#define S_LEN 2048
#define B_SZ  2
#define H_DIM 2048
#define NHEAD 16
#define HDIM  128
#define K_DIM 2048

typedef __attribute__((ext_vector_type(16))) __bf16 v16bf;
typedef __attribute__((ext_vector_type(8)))  __bf16 v8bf;
typedef __attribute__((ext_vector_type(2)))  __bf16 v2bf;
typedef __attribute__((ext_vector_type(8)))  float  v8f;
typedef __attribute__((ext_vector_type(4)))  unsigned v4u;
typedef __attribute__((ext_vector_type(8)))  int    v8i;
typedef __attribute__((ext_vector_type(4)))  int    v4i;

#if __has_builtin(__builtin_amdgcn_tensor_load_to_lds) && \
    __has_builtin(__builtin_amdgcn_s_wait_tensorcnt)
#define USE_TDM 1
#endif

__device__ __forceinline__ __bf16 f2bf(float f) { return (__bf16)f; }

// pack two f32 -> packed bf16 pair (one v_cvt_pk_bf16_f32 when available)
__device__ __forceinline__ unsigned pkbf(float a, float b) {
#if __has_builtin(__builtin_amdgcn_cvt_pk_bf16_f32)
  v2bf t = __builtin_amdgcn_cvt_pk_bf16_f32(a, b);
  return __builtin_bit_cast(unsigned, t);
#else
  unsigned short l = __builtin_bit_cast(unsigned short, (__bf16)a);
  unsigned short h = __builtin_bit_cast(unsigned short, (__bf16)b);
  return (unsigned)l | ((unsigned)h << 16);
#endif
}

__device__ __forceinline__ float fexp2(float x) {
#if __has_builtin(__builtin_amdgcn_exp2f)
  return __builtin_amdgcn_exp2f(x);
#else
  return exp2f(x);
#endif
}

__device__ __forceinline__ v16bf ld_frag(const __bf16* p0, const __bf16* p1) {
  v8bf lo = *(const v8bf*)p0;
  v8bf hi = *(const v8bf*)p1;
  return __builtin_shufflevector(lo, hi, 0,1,2,3,4,5,6,7,8,9,10,11,12,13,14,15);
}

__device__ __forceinline__ v8f wmma_bf16(v16bf a, v16bf b, v8f c) {
  return __builtin_amdgcn_wmma_f32_16x16x32_bf16(false, a, false, b, (short)0, c,
                                                 false, false);
}

__device__ __forceinline__ v8f vzero8() {
  v8f z = {0.f,0.f,0.f,0.f,0.f,0.f,0.f,0.f};
  return z;
}

#ifdef USE_TDM
// Issue one 2-D TDM tile load (D# per CDNA5 ISA sec 8.3/8.4; groups 2/3 zero).
// g1 words: w0 = flags (data_size / pad), w1..w4 = dims/tiles, w5 = stride lo.
// 6-arg builtin form (clang-23 / therock-10.0 headers).
__device__ __forceinline__ void tdm_load_2d(unsigned ldsaddr, unsigned long long ga,
                                            int w0, int w1, int w2, int w3,
                                            int w4, int w5) {
  v4u g0;
  g0[0] = 1u;                                  // count=1, user mode
  g0[1] = ldsaddr;                             // lds_addr (bytes)
  g0[2] = (unsigned)(ga & 0xFFFFFFFFu);        // global_addr[31:0]
  g0[3] = (unsigned)((ga >> 32) & 0x1FFFFFFu)  // global_addr[56:32]
          | (2u << 30);                        // type = 2 ("image")
  v8i g1;
  g1[0] = w0; g1[1] = w1; g1[2] = w2; g1[3] = w3;
  g1[4] = w4; g1[5] = w5; g1[6] = 0; g1[7] = 0;
  v4i gz4 = {0, 0, 0, 0};
  v8i gz8 = {0, 0, 0, 0, 0, 0, 0, 0};
  __builtin_amdgcn_tensor_load_to_lds(g0, g1, gz4, gz4, gz8, 0);
}
#endif

// ---------------------------------------------------------------------------
// Kernel 1: QKV projection. C[4096,6144] = hidden[4096,2048] * w_qkv^T + b_qkv
// scattered as bf16 into Q/K/V buffers laid out [b][head][s][d].
// ---------------------------------------------------------------------------
__global__ __launch_bounds__(256) void qkv_gemm(
    const float* __restrict__ A, const float* __restrict__ W,
    const float* __restrict__ bias,
    __bf16* __restrict__ Qb, __bf16* __restrict__ Kb, __bf16* __restrict__ Vb) {
  __shared__ __bf16 As[128 * 40];   // stride 40 elems = 80B (16B aligned rows)
  __shared__ __bf16 Bs[128 * 40];
  const int tid   = threadIdx.x;
  const int lane  = tid & 31, w = tid >> 5;
  const int laneN = lane & 15, laneH = lane >> 4;
  const int wm = w >> 1, wn = w & 1;           // 4x2 wave grid
  const int mb = blockIdx.y, nb = blockIdx.x;

  v8f acc[2][4];
  for (int i = 0; i < 2; ++i)
    for (int j = 0; j < 4; ++j) acc[i][j] = vzero8();

  const int rowb = tid >> 3;        // 0..31
  const int kq   = (tid & 7) * 4;   // 0..28 step 4

  for (int kt = 0; kt < K_DIM; kt += 32) {
    for (int rr = 0; rr < 4; ++rr) {
      int m = rowb + 32 * rr;
      float4 v = *(const float4*)(A + (size_t)(mb * 128 + m) * K_DIM + kt + kq);
      uint2 p; p.x = pkbf(v.x, v.y); p.y = pkbf(v.z, v.w);
      *(uint2*)&As[m * 40 + kq] = p;
    }
    for (int rr = 0; rr < 4; ++rr) {
      int n = rowb + 32 * rr;
      float4 v = *(const float4*)(W + (size_t)(nb * 128 + n) * K_DIM + kt + kq);
      uint2 p; p.x = pkbf(v.x, v.y); p.y = pkbf(v.z, v.w);
      *(uint2*)&Bs[n * 40 + kq] = p;
    }
    if (kt + 32 < K_DIM) {
      __builtin_prefetch(A + (size_t)(mb * 128 + rowb) * K_DIM + kt + 32 + kq, 0, 1);
      __builtin_prefetch(W + (size_t)(nb * 128 + rowb) * K_DIM + kt + 32 + kq, 0, 1);
    }
    __syncthreads();
    v16bf af[2], bfr[4];
    for (int im = 0; im < 2; ++im) {
      const __bf16* ap = &As[(wm * 32 + im * 16 + laneN) * 40 + 8 * laneH];
      af[im] = ld_frag(ap, ap + 16);
    }
    for (int in_ = 0; in_ < 4; ++in_) {
      const __bf16* bp = &Bs[(wn * 64 + in_ * 16 + laneN) * 40 + 8 * laneH];
      bfr[in_] = ld_frag(bp, bp + 16);
    }
    for (int im = 0; im < 2; ++im)
      for (int in_ = 0; in_ < 4; ++in_)
        acc[im][in_] = wmma_bf16(af[im], bfr[in_], acc[im][in_]);
    __syncthreads();
  }

  // epilogue: bias + scatter to per-head Q/K/V (bf16)
  for (int in_ = 0; in_ < 4; ++in_) {
    int n_g   = nb * 128 + wn * 64 + in_ * 16 + laneN;
    float bv  = bias[n_g];
    int head  = n_g / 384;
    int inner = n_g % 384;
    int which = inner >> 7;
    int dI    = inner & 127;
    __bf16* buf = (which == 0) ? Qb : ((which == 1) ? Kb : Vb);
    for (int im = 0; im < 2; ++im) {
      v8f c = acc[im][in_];
      for (int r = 0; r < 8; ++r) {
        int m_g = mb * 128 + wm * 32 + im * 16 + r + 8 * laneH;
        int sI = m_g >> 1, bI = m_g & 1;
        buf[(size_t)(((bI * NHEAD + head) * S_LEN) + sI) * HDIM + dI] =
            f2bf(c[r] + bv);
      }
    }
  }
}

// ---------------------------------------------------------------------------
// Kernel 2: causal flash attention, one (b, head, 128-row q block) per block.
// 8 waves x 16 q-rows. K tile = 64 keys per iteration (staged by the TDM).
// ---------------------------------------------------------------------------
__global__ __launch_bounds__(256) void attn_kernel(
    const __bf16* __restrict__ Qb, const __bf16* __restrict__ Kb,
    const __bf16* __restrict__ Vb, __bf16* __restrict__ Cb) {
  __shared__ __bf16 Ks[64 * 136];    // [key][feat], stride 136 (272B rows)
  __shared__ __bf16 VsT[128 * 72];   // [feat][key], stride 72 (144B rows)
  __shared__ __bf16 Ps[8 * 16 * 72]; // per-wave P staging (D->A relayout)
  const int tid   = threadIdx.x;
  const int lane  = tid & 31, w = tid >> 5;
  const int laneN = lane & 15, laneH = lane >> 4;
  const int qb = blockIdx.x, head = blockIdx.y, b = blockIdx.z;

  const size_t hb = (size_t)((b * NHEAD + head) * S_LEN) * HDIM;
  const __bf16* Qh = Qb + hb;
  const __bf16* Kh = Kb + hb;
  const __bf16* Vh = Vb + hb;

  // Q fragments (A layout), resident in registers for the whole block
  const int qrow = qb * 128 + w * 16 + laneN;
  const __bf16* qp = Qh + (size_t)qrow * HDIM;
  v16bf qa[4];
  for (int st = 0; st < 4; ++st) {
    int base = st * 32 + 8 * laneH;
    qa[st] = ld_frag(qp + base, qp + base + 16);
  }

  float mrow[8], lrow[8];
  v8f ctx[8];
  for (int r = 0; r < 8; ++r) { mrow[r] = -3.0e38f; lrow[r] = 0.f; }
  for (int nt = 0; nt < 8; ++nt) ctx[nt] = vzero8();

  const float SC = 0.08838834764831845f * 1.44269504088896340f; // 1/sqrt(128)*log2e
  const int kbmax = 2 * qb + 1;
  for (int kb = 0; kb <= kbmax; ++kb) {
#ifdef USE_TDM
    // stage K [key][feat] via Tensor Data Mover: 64x128 bf16 tile,
    // LDS row padded 256B data + 16B (pad_interval=5 -> 64 DW, pad_amount=3 -> 4 DW)
    if (w == 0) {
      tdm_load_2d((unsigned)(size_t)&Ks[0],
                  (unsigned long long)(size_t)(Kh + (size_t)(kb * 64) * HDIM),
                  0x00010000 | (1 << 20) | (5 << 22) | (3 << 25), // 2B, pad on
                  (int)(128u << 16),   // tensor_dim0 = 128
                  (int)(2048u << 16),  // tensor_dim1 = 2048
                  (int)(128u << 16),   // tile_dim0 = 128
                  64,                  // tile_dim1 = 64
                  128);                // tensor_dim0_stride = 128
    }
#else
    for (int idx = tid; idx < 64 * 16; idx += 256) {
      int row = idx >> 4, ch = idx & 15;
      *(uint4*)&Ks[row * 136 + ch * 8] =
          *(const uint4*)(Kh + (size_t)(kb * 64 + row) * HDIM + ch * 8);
    }
#endif
    // stage V transposed [feat][key]
    for (int idx = tid; idx < 64 * 64; idx += 256) {
      int key = idx >> 6, fp = (idx & 63) * 2;
      unsigned u = *(const unsigned*)(Vh + (size_t)(kb * 64 + key) * HDIM + fp);
      VsT[fp * 72 + key]       = __builtin_bit_cast(__bf16, (unsigned short)(u & 0xFFFFu));
      VsT[(fp + 1) * 72 + key] = __builtin_bit_cast(__bf16, (unsigned short)(u >> 16));
    }
#ifdef USE_TDM
    if (w == 0) __builtin_amdgcn_s_wait_tensorcnt(0);
#endif
    __syncthreads();

    // S = Q * K^T (16 q-rows x 64 keys per wave)
    v8f sacc[4];
    for (int t = 0; t < 4; ++t) sacc[t] = vzero8();
    for (int st = 0; st < 4; ++st) {
      for (int t = 0; t < 4; ++t) {
        const __bf16* kp = &Ks[(t * 16 + laneN) * 136 + st * 32 + 8 * laneH];
        sacc[t] = wmma_bf16(qa[st], ld_frag(kp, kp + 16), sacc[t]);
      }
    }

    // causal mask + online softmax in exp2 domain
    float tv[4][8];
    for (int t = 0; t < 4; ++t) {
      int kg = kb * 64 + t * 16 + laneN;
      for (int r = 0; r < 8; ++r) {
        int qg = qb * 128 + w * 16 + r + 8 * laneH;
        float s = sacc[t][r] * SC;
        tv[t][r] = (kg <= qg) ? s : -3.0e38f;
      }
    }
    float alpha[8];
    for (int r = 0; r < 8; ++r) {
      float vm = fmaxf(fmaxf(tv[0][r], tv[1][r]), fmaxf(tv[2][r], tv[3][r]));
      for (int msk = 1; msk < 16; msk <<= 1) vm = fmaxf(vm, __shfl_xor(vm, msk, 32));
      float mn = fmaxf(mrow[r], vm);
      alpha[r] = fexp2(mrow[r] - mn);
      float rs = 0.f;
      for (int t = 0; t < 4; ++t) { float p = fexp2(tv[t][r] - mn); tv[t][r] = p; rs += p; }
      for (int msk = 1; msk < 16; msk <<= 1) rs += __shfl_xor(rs, msk, 32);
      lrow[r] = lrow[r] * alpha[r] + rs;
      mrow[r] = mn;
    }
    for (int nt = 0; nt < 8; ++nt)
      for (int r = 0; r < 8; ++r) ctx[nt][r] *= alpha[r];

    // relayout P (D layout -> A layout) through per-wave LDS slab
    __bf16* pw = &Ps[w * 16 * 72];
    for (int t = 0; t < 4; ++t)
      for (int r = 0; r < 8; ++r)
        pw[(r + 8 * laneH) * 72 + t * 16 + laneN] = f2bf(tv[t][r]);

    // ctx += P * V
    for (int ksb = 0; ksb < 2; ++ksb) {
      const __bf16* pp = pw + laneN * 72 + ksb * 32 + 8 * laneH;
      v16bf pa = ld_frag(pp, pp + 16);
      for (int nt = 0; nt < 8; ++nt) {
        const __bf16* vp = &VsT[(nt * 16 + laneN) * 72 + ksb * 32 + 8 * laneH];
        ctx[nt] = wmma_bf16(pa, ld_frag(vp, vp + 16), ctx[nt]);
      }
    }
    __syncthreads();
  }

  for (int r = 0; r < 8; ++r) lrow[r] = 1.0f / lrow[r];
  const int srow_base = qb * 128 + w * 16;
  for (int nt = 0; nt < 8; ++nt) {
    int feat = head * HDIM + nt * 16 + laneN;
    for (int r = 0; r < 8; ++r) {
      int srow = srow_base + r + 8 * laneH;
      Cb[(size_t)(srow * B_SZ + b) * H_DIM + feat] = f2bf(ctx[nt][r] * lrow[r]);
    }
  }
}

// ---------------------------------------------------------------------------
// Kernel 3: dense projection. out[4096,2048] = ctx_bf16 * w_dense^T (f32 out)
// A tile (already bf16) staged by the TDM; B tile converted in-flight.
// ---------------------------------------------------------------------------
__global__ __launch_bounds__(256) void dense_gemm(
    const __bf16* __restrict__ A, const float* __restrict__ W,
    float* __restrict__ out) {
  __shared__ __bf16 As[128 * 40];
  __shared__ __bf16 Bs[128 * 40];
  const int tid   = threadIdx.x;
  const int lane  = tid & 31, w = tid >> 5;
  const int laneN = lane & 15, laneH = lane >> 4;
  const int wm = w >> 1, wn = w & 1;
  const int mb = blockIdx.y, nb = blockIdx.x;

  v8f acc[2][4];
  for (int i = 0; i < 2; ++i)
    for (int j = 0; j < 4; ++j) acc[i][j] = vzero8();

  const int rowbA = tid >> 2;        // 0..63
  const int kqA   = (tid & 3) * 8;   // bf16 elems
  const int rowbB = tid >> 3;        // 0..31
  const int kqB   = (tid & 7) * 4;   // f32 elems

  for (int kt = 0; kt < K_DIM; kt += 32) {
#ifdef USE_TDM
    // 128x32 bf16 tile; LDS row = 64B data + 16B pad (interval=2 -> 8 DW, amount=3)
    if (w == 0) {
      tdm_load_2d((unsigned)(size_t)&As[0],
                  (unsigned long long)(size_t)(A + (size_t)(mb * 128) * K_DIM + kt),
                  0x00010000 | (1 << 20) | (2 << 22) | (3 << 25),
                  (int)((K_DIM & 0xFFFF) << 16),               // tensor_dim0 lo16
                  (int)((K_DIM >> 16) | ((4096u & 0xFFFF) << 16)), // dim0 hi / dim1 lo
                  (int)((4096u >> 16) | (32u << 16)),          // dim1 hi / tile_dim0=32
                  128,                                         // tile_dim1 = 128
                  K_DIM);                                      // tensor_dim0_stride
    }
#else
    for (int rr = 0; rr < 2; ++rr) {
      int m = rowbA + 64 * rr;
      *(uint4*)&As[m * 40 + kqA] =
          *(const uint4*)(A + (size_t)(mb * 128 + m) * K_DIM + kt + kqA);
    }
#endif
    for (int rr = 0; rr < 4; ++rr) {
      int n = rowbB + 32 * rr;
      float4 v = *(const float4*)(W + (size_t)(nb * 128 + n) * K_DIM + kt + kqB);
      uint2 p; p.x = pkbf(v.x, v.y); p.y = pkbf(v.z, v.w);
      *(uint2*)&Bs[n * 40 + kqB] = p;
    }
    if (kt + 32 < K_DIM) {
      __builtin_prefetch(A + (size_t)(mb * 128 + rowbA) * K_DIM + kt + 32 + kqA, 0, 1);
      __builtin_prefetch(W + (size_t)(nb * 128 + rowbB) * K_DIM + kt + 32 + kqB, 0, 1);
    }
#ifdef USE_TDM
    if (w == 0) __builtin_amdgcn_s_wait_tensorcnt(0);
#endif
    __syncthreads();
    v16bf af[2], bfr[4];
    for (int im = 0; im < 2; ++im) {
      const __bf16* ap = &As[(wm * 32 + im * 16 + laneN) * 40 + 8 * laneH];
      af[im] = ld_frag(ap, ap + 16);
    }
    for (int in_ = 0; in_ < 4; ++in_) {
      const __bf16* bp = &Bs[(wn * 64 + in_ * 16 + laneN) * 40 + 8 * laneH];
      bfr[in_] = ld_frag(bp, bp + 16);
    }
    for (int im = 0; im < 2; ++im)
      for (int in_ = 0; in_ < 4; ++in_)
        acc[im][in_] = wmma_bf16(af[im], bfr[in_], acc[im][in_]);
    __syncthreads();
  }

  for (int in_ = 0; in_ < 4; ++in_) {
    int n_g = nb * 128 + wn * 64 + in_ * 16 + laneN;
    for (int im = 0; im < 2; ++im) {
      v8f c = acc[im][in_];
      for (int r = 0; r < 8; ++r) {
        int m_g = mb * 128 + wm * 32 + im * 16 + r + 8 * laneH;
        out[(size_t)m_g * H_DIM + n_g] = c[r];
      }
    }
  }
}

__global__ void bias_copy(const float* __restrict__ b, float* __restrict__ out) {
  int i = blockIdx.x * 256 + threadIdx.x;
  if (i < H_DIM) out[i] = b[i];
}

// ---------------------------------------------------------------------------
extern "C" void kernel_launch(void* const* d_in, const int* in_sizes, int n_in,
                              void* d_out, int out_size, void* d_ws, size_t ws_size,
                              hipStream_t stream) {
  (void)in_sizes; (void)n_in; (void)out_size; (void)ws_size;
  const float* hidden  = (const float*)d_in[0];
  // d_in[1] = attention_mask (causal; computed analytically, unused)
  const float* w_qkv   = (const float*)d_in[2];
  const float* b_qkv   = (const float*)d_in[3];
  const float* w_dense = (const float*)d_in[4];
  const float* b_dense = (const float*)d_in[5];
  float* out = (float*)d_out;

  const size_t perbuf = (size_t)B_SZ * NHEAD * S_LEN * HDIM;  // 8,388,608 elems
  __bf16* Qb = (__bf16*)d_ws;
  __bf16* Kb = Qb + perbuf;
  __bf16* Vb = Kb + perbuf;
  __bf16* Cb = Vb + perbuf;   // ctx, [s*B+b][h], bf16   (total ws: 64 MB)

  qkv_gemm<<<dim3(48, 32), 256, 0, stream>>>(hidden, w_qkv, b_qkv, Qb, Kb, Vb);
  attn_kernel<<<dim3(16, NHEAD, B_SZ), 256, 0, stream>>>(Qb, Kb, Vb, Cb);
  dense_gemm<<<dim3(16, 32), 256, 0, stream>>>(Cb, w_dense, out);
  bias_copy<<<dim3(8), 256, 0, stream>>>(b_dense, out + (size_t)S_LEN * B_SZ * H_DIM);
}
